// SpatialCompatibilityFeatures_51926154609365
// MI455X (gfx1250) — matrix operand order repaired and checked by reading, not compiled, not run on hardware
//
#include <hip/hip_runtime.h>
#include <cmath>

typedef _Float16 v16h __attribute__((ext_vector_type(16)));
typedef _Float16 v8h  __attribute__((ext_vector_type(8)));
typedef float    v8f  __attribute__((ext_vector_type(8)));

#define Bq 4
#define Nq 512
#define Hq 128

// Branchless erf-based GELU: Abramowitz & Stegun 7.1.26 (|err| < 1.5e-7),
// using native v_rcp_f32 / v_exp_f32 so no divergent EXEC regions are
// generated between WMMA ops.
__device__ __forceinline__ float gelu_exactish(float x) {
    const float a = fabsf(x) * 0.70710678118654752f;   // |x|/sqrt(2)
    const float t = __builtin_amdgcn_rcpf(fmaf(0.3275911f, a, 1.0f));
    float p = fmaf(t, 1.061405429f, -1.453152027f);
    p = fmaf(t, p, 1.421413741f);
    p = fmaf(t, p, -0.284496736f);
    p = fmaf(t, p, 0.254829592f);
    p = p * t;
    const float e    = __expf(-a * a);                 // native exp, branchless
    const float erfa = fmaf(-p, e, 1.0f);              // erf(a) in [0,1]
    const float erfx = copysignf(erfa, x);             // erf(x/sqrt(2))
    return 0.5f * x * (1.0f + erfx);
}

// Each block: 128 threads = 4 waves. Each wave handles 16 (m,n) pairs
// (one m, 16 consecutive n) and produces the full 128-wide MLP output
// for those pairs via WMMA.
__global__ __launch_bounds__(128) void scf_mlp_kernel(
    const float* __restrict__ bbox_i, const float* __restrict__ bbox_j,
    const float* __restrict__ W1, const float* __restrict__ b1,
    const float* __restrict__ W2, const float* __restrict__ b2,
    float* __restrict__ out)
{
    // W1 staged as [j][kk] : kk = 0..31 (K padded 18->32), column j = output unit.
    __shared__ __align__(32) _Float16 sW1[32 * Hq];
    // W2 staged as [c][j][kk] : c = K-chunk (0..3), kk = k within chunk.
    __shared__ __align__(32) _Float16 sW2[4 * Hq * 32];
    __shared__ float sB1[Hq];
    __shared__ float sB2[Hq];
    // Per-wave hidden activations, row-major [pair_row][col], f16.
    __shared__ __align__(32) _Float16 sH[4][16 * Hq];

    const int tid = threadIdx.x;

    // ---- Stage weights into LDS in WMMA B-operand-friendly layout ----
    for (int idx = tid; idx < 32 * Hq; idx += 128) {
        int j = idx >> 5, kk = idx & 31;
        float v = (kk < 18) ? W1[kk * Hq + j] : 0.0f;
        sW1[idx] = (_Float16)v;
    }
    for (int idx = tid; idx < 4 * Hq * 32; idx += 128) {
        int c = idx >> 12, j = (idx >> 5) & 127, kk = idx & 31;
        sW2[idx] = (_Float16)W2[(c * 32 + kk) * Hq + j];
    }
    if (tid < Hq) { sB1[tid] = b1[tid]; sB2[tid] = b2[tid]; }
    __syncthreads();

    const int lane = tid & 31;
    const int wave = tid >> 5;
    const int col  = lane & 15;   // N-index inside 16x16 tile; also this lane's A-row
    const int half = lane >> 4;   // which K-half of the fragment this lane holds

    const long long pairBase = (long long)blockIdx.x * 64 + wave * 16;
    const int b  = (int)(pairBase / (Nq * Nq));
    const int rm = (int)(pairBase % (Nq * Nq));
    const int m  = rm / Nq;
    const int n  = (rm % Nq) + col;

    // ---- Spatial features (each lane: pair row = col) ----
    const float4 bi = *(const float4*)(bbox_i + ((size_t)b * Nq + m) * 4);
    const float4 bj = *(const float4*)(bbox_j + ((size_t)b * Nq + n) * 4);
    const float x1i = bi.x, y1i = bi.y, x2i = bi.z, y2i = bi.w;
    const float x1j = bj.x, y1j = bj.y, x2j = bj.z, y2j = bj.w;
    const float wi = x2i - x1i, hi = y2i - y1i;
    const float wj = x2j - x1j, hj = y2j - y1j;
    const float cxi = (x1i + x2i) * 0.5f, cyi = (y1i + y2i) * 0.5f;
    const float cxj = (x1j + x2j) * 0.5f, cyj = (y1j + y2j) * 0.5f;
    const float dcx = cxj - cxi, dcy = cyj - cyi;

    float f[18];
    f[0]  = dcx;
    f[1]  = dcy;
    f[2]  = x1j - x1i;
    f[3]  = y1j - y1i;
    f[4]  = x2j - x2i;
    f[5]  = y2j - y2i;
    // log(a/b) = log(a) - log(b): branchless fast logs, no f32 division.
    f[6]  = __logf(wj + 1.0f) - __logf(wi + 1.0f);
    f[7]  = __logf(hj + 1.0f) - __logf(hi + 1.0f);
    f[8]  = fmaxf(fminf(x2i, x2j) - fmaxf(x1i, x1j), 0.0f);
    f[9]  = fmaxf(fminf(y2i, y2j) - fmaxf(y1i, y1j), 0.0f);
    f[10] = __builtin_amdgcn_sqrtf(fmaf(dcx, dcx, fmaf(dcy, dcy, 1e-6f)));
    f[11] = fabsf(dcx);
    f[12] = fabsf(dcy);
    f[13] = (cxj > cxi) ? 1.0f : 0.0f;
    f[14] = (cyj > cyi) ? 1.0f : 0.0f;
    f[15] = (x1j > x2i) ? 1.0f : 0.0f;
    f[16] = (y1j > y2i) ? 1.0f : 0.0f;
    f[17] = __logf(fmaf(wj, hj, 1.0f)) - __logf(fmaf(wi, hi, 1.0f));

    // ---- Pack A fragment for layer-1 WMMA (16x32 f16, K padded) ----
    // A layout: lanes 0-15 hold K {e} for elements e=0..15;
    //           lanes 16-31 hold those K values +16.
    v16h af;
#pragma unroll
    for (int e = 0; e < 16; e++) {
        const int k0 = e;                       // half == 0
        const int k1 = k0 + 16;                 // half == 1
        const float v0 = (k0 < 18) ? f[k0] : 0.0f;
        const float v1 = (k1 < 18) ? f[k1] : 0.0f;
        af[e] = (_Float16)(half ? v1 : v0);
    }

    // ---- Layer 1: h = gelu(f @ W1 + b1), staged to LDS in f16 ----
    _Float16* hbuf = sH[wave];
#pragma unroll
    for (int t = 0; t < 8; t++) {
        const int j = t * 16 + col;
        const v16h bw = *(const v16h*)(sW1 + j * 32 + half * 16);
        v8f c1 = {};
        c1 = __builtin_amdgcn_wmma_f32_16x16x32_f16(
                 false, af, false, bw, (short)0, c1, false, false);
        const float bias = sB1[j];
#pragma unroll
        for (int r = 0; r < 8; r++) {
            const float g = gelu_exactish(c1[r] + bias);
            hbuf[(r + 8 * half) * Hq + j] = (_Float16)g;
        }
    }
    __syncthreads();

    // ---- Reload h as A fragments for layer 2 (K = 128 => 4 chunks) ----
    v16h a2[4];
#pragma unroll
    for (int c = 0; c < 4; c++) {
        const v8h lo = *(const v8h*)(hbuf + col * Hq + c * 32 + 16 * half);
        const v8h hh = *(const v8h*)(hbuf + col * Hq + c * 32 + 16 * half + 8);
        v16h a;
#pragma unroll
        for (int i = 0; i < 8; i++) { a[i] = lo[i]; a[i + 8] = hh[i]; }
        a2[c] = a;
    }

    // ---- Layer 2: out = h @ W2 + b2, streamed to global ----
#pragma unroll
    for (int t = 0; t < 8; t++) {
        const int j = t * 16 + col;
        v8f acc = {};
#pragma unroll
        for (int c = 0; c < 4; c++) {
            const v16h bw = *(const v16h*)(sW2 + (c * Hq + j) * 32 + 16 * half);
            acc = __builtin_amdgcn_wmma_f32_16x16x32_f16(
                      false, a2[c], false, bw, (short)0, acc, false, false);
        }
        const float bias2 = sB2[j];
#pragma unroll
        for (int r = 0; r < 8; r++) {
            const size_t pr = (size_t)(pairBase + r + 8 * half);
            out[pr * Hq + j] = acc[r] + bias2;
        }
    }
}

extern "C" void kernel_launch(void* const* d_in, const int* in_sizes, int n_in,
                              void* d_out, int out_size, void* d_ws, size_t ws_size,
                              hipStream_t stream) {
    (void)in_sizes; (void)n_in; (void)out_size; (void)d_ws; (void)ws_size;
    const float* bbox_i = (const float*)d_in[0];
    const float* bbox_j = (const float*)d_in[1];
    const float* W1     = (const float*)d_in[2];
    const float* b1     = (const float*)d_in[3];
    const float* W2     = (const float*)d_in[4];
    const float* b2     = (const float*)d_in[5];
    float* out = (float*)d_out;

    const int totalPairs = Bq * Nq * Nq;          // 1,048,576
    dim3 grid(totalPairs / 64);                   // 64 pairs per block
    dim3 block(128);                              // 4 waves
    scf_mlp_kernel<<<grid, block, 0, stream>>>(bbox_i, bbox_j, W1, b1, W2, b2, out);
}
